// AttentionHead_36747740185062
// MI455X (gfx1250) — matrix operand order repaired and checked by reading, not compiled, not run on hardware
//
#include <hip/hip_runtime.h>
#include <hip/hip_bf16.h>

// ---------------------------------------------------------------------------
// CDNA5 (gfx1250) fused attention head: QKV projection GEMMs + flash attention
// v_wmma_f32_16x16x32_bf16 everywhere; double-buffered async global->LDS
// staging; software-pipelined LDS fragment loads behind the WMMA stream.
// ---------------------------------------------------------------------------

#define BATCH 8
#define SEQ   2048
#define EMB   1024
#define DHEAD 64

typedef __attribute__((ext_vector_type(16))) __bf16 v16bf;
typedef __attribute__((ext_vector_type(8)))  float  v8f;
typedef __attribute__((ext_vector_type(4)))  int    v4i;

union ABf {
    v16bf v;
    unsigned short u[16];
    unsigned int   d[8];
    uint4 q[2];
};

// fp32 -> bf16, round-to-nearest at bit 16 (cheap: add + shift)
__device__ __forceinline__ unsigned short f2bf(float f) {
    return (unsigned short)((__float_as_uint(f) + 0x8000u) >> 16);
}

// two fp32 -> packed bf16x2: 2 adds + 1 v_perm_b32
__device__ __forceinline__ unsigned int f2bf2(float lo, float hi) {
    unsigned int a = __float_as_uint(lo) + 0x8000u;   // src1
    unsigned int b = __float_as_uint(hi) + 0x8000u;   // src0
    return __builtin_amdgcn_perm(b, a, 0x07060302u);  // {b[31:16], a[31:16]}
}

// ---- async global -> LDS (gfx1250 path, guarded), 16B granules -------------
#if defined(__HIP_DEVICE_COMPILE__) && \
    __has_builtin(__builtin_amdgcn_global_load_async_to_lds_b128) && \
    __has_builtin(__builtin_amdgcn_s_wait_asynccnt)
#define HAVE_ASYNC_LDS 1
#endif

__device__ __forceinline__ void cp16_g2l(const unsigned short* g, unsigned short* l) {
#ifdef HAVE_ASYNC_LDS
    typedef __attribute__((address_space(1))) v4i g4i_t;   // global int4
    typedef __attribute__((address_space(3))) v4i l4i_t;   // LDS int4
    __builtin_amdgcn_global_load_async_to_lds_b128((g4i_t*)g, (l4i_t*)l, 0, 0);
#else
    *(uint4*)l = *(const uint4*)g;
#endif
}

__device__ __forceinline__ void async_wait() {
#ifdef HAVE_ASYNC_LDS
    __builtin_amdgcn_s_wait_asynccnt(0);
#endif
}

__device__ __forceinline__ v8f wmma_bf16(const v16bf& a, const v16bf& b, const v8f& c) {
    return __builtin_amdgcn_wmma_f32_16x16x32_bf16(
        false, a, false, b, (short)0, c, false, false);
}

// ---------------------------------------------------------------------------
// Kernel 1: swizzle W (E x D f32) -> bf16 per-lane B-fragment order.
// Layout: [kt(32)][mn(12)=m*4+nt][lane(32)][elem(16)]  (one 12 KB slab per kt)
// Lane elem e:  k = kt*32 + 8*(lane/16) + (e%8) + 16*(e/8),  n = nt*16 + lane%16.
// ---------------------------------------------------------------------------
__global__ __launch_bounds__(256) void wprep_kernel(
    const float* __restrict__ Wq, const float* __restrict__ Wk,
    const float* __restrict__ Wv, unsigned short* __restrict__ wswz)
{
    int t = blockIdx.x * 256 + threadIdx.x;        // 32*12*512 = 196608
    if (t >= 32 * 12 * 512) return;
    int e    =  t       & 15;
    int lane = (t >> 4) & 31;
    int rest =  t >> 9;
    int mn   = rest % 12;
    int kt   = rest / 12;
    int m    = mn >> 2;
    int nt   = mn & 3;
    int k = kt * 32 + ((lane >> 4) << 3) + (e & 7) + ((e >> 3) << 4);
    int n = nt * 16 + (lane & 15);
    const float* W = (m == 0) ? Wq : ((m == 1) ? Wk : Wv);
    wswz[t] = f2bf(W[(size_t)k * DHEAD + n]);
}

// ---------------------------------------------------------------------------
// Kernel 2: Q/K/V projections. 128 blocks x 256 threads (8 waves).
// Weight-fragment slab per K-step shared via double-buffered async LDS;
// B fragments pipelined (depth 2) behind the 12-WMMA stream.
// ---------------------------------------------------------------------------
__global__ __launch_bounds__(256) void qkv_kernel(
    const float* __restrict__ x, const unsigned short* __restrict__ wswz,
    unsigned short* __restrict__ qbf, unsigned short* __restrict__ kbf,
    unsigned short* __restrict__ vt)
{
    __shared__ unsigned short Blds[2][12 * 512];   // 2 x 12 KB
    const int tid  = threadIdx.x;
    const int lane = tid & 31;
    const int wave = tid >> 5;
    const int rowbase = blockIdx.x * 128 + wave * 16;
    const int h  = lane >> 4;
    const int ln = lane & 15;
    const int arow = rowbase + ln;

    v8f acc[12];
    #pragma unroll
    for (int i = 0; i < 12; ++i)
        #pragma unroll
        for (int v = 0; v < 8; ++v) acc[i][v] = 0.0f;

    // stage slab for kt into LDS buffer (cooperative: 768 x 16B, 3 per thread)
    auto stage = [&](int kt, int buf) {
        const unsigned short* src = wswz + (size_t)kt * 6144;
        unsigned short* dst = &Blds[buf][0];
        #pragma unroll
        for (int r = 0; r < 3; ++r) {
            int i = r * 256 + tid;
            cp16_g2l(src + i * 8, dst + i * 8);
        }
    };

    stage(0, 0);
    int cur = 0;
    for (int kt = 0; kt < EMB / 32; ++kt) {
        async_wait();
        __syncthreads();
        if (kt + 1 < EMB / 32) stage(kt + 1, cur ^ 1);

        // ---- A fragment: 16x32 from x, f32 -> bf16 (add + v_perm packs) ----
        const float* xp = x + (size_t)arow * EMB + kt * 32 + h * 8;
        float tf[16];
        *(float4*)(tf + 0)  = *(const float4*)(xp + 0);
        *(float4*)(tf + 4)  = *(const float4*)(xp + 4);
        *(float4*)(tf + 8)  = *(const float4*)(xp + 16);
        *(float4*)(tf + 12) = *(const float4*)(xp + 20);
        ABf A;
        #pragma unroll
        for (int e = 0; e < 8; ++e) A.d[e] = f2bf2(tf[2 * e], tf[2 * e + 1]);

        // ---- depth-2 pipelined B fragments from LDS + 12 WMMAs ----
        const unsigned short* bl = &Blds[cur][0] + lane * 16;
        ABf Bf[3];
        Bf[0].q[0] = *(const uint4*)(bl + 0);
        Bf[0].q[1] = *(const uint4*)(bl + 8);
        Bf[1].q[0] = *(const uint4*)(bl + 512);
        Bf[1].q[1] = *(const uint4*)(bl + 520);
        #pragma unroll
        for (int mn = 0; mn < 12; ++mn) {
            if (mn + 2 < 12) {
                Bf[(mn + 2) % 3].q[0] = *(const uint4*)(bl + (mn + 2) * 512);
                Bf[(mn + 2) % 3].q[1] = *(const uint4*)(bl + (mn + 2) * 512 + 8);
            }
            acc[mn] = wmma_bf16(A.v, Bf[mn % 3].v, acc[mn]);
        }

        cur ^= 1;
    }

    // ---- epilogue: Q,K row-major bf16; V transposed bf16 (packed stores) ----
    #pragma unroll
    for (int nt = 0; nt < 4; ++nt) {
        int col = nt * 16 + ln;
        #pragma unroll
        for (int v = 0; v < 8; ++v) {
            int srow = rowbase + v + 8 * h;
            qbf[(size_t)srow * DHEAD + col] = f2bf(acc[0 + nt][v]);
            kbf[(size_t)srow * DHEAD + col] = f2bf(acc[4 + nt][v]);
        }
        int b  = rowbase >> 11;                 // whole tile in one batch
        int sl = (rowbase & 2047) + 8 * h;
        #pragma unroll
        for (int v = 0; v < 8; v += 2) {        // packed 2 x bf16 stores
            unsigned int pk = f2bf2(acc[8 + nt][v], acc[8 + nt][v + 1]);
            *(unsigned int*)(vt + ((size_t)b * DHEAD + col) * SEQ + sl + v) = pk;
        }
    }
}

// ---------------------------------------------------------------------------
// Kernel 3: flash attention. 128 blocks x 256 threads; one wave per 16-row
// query tile (8 tiles of one batch per block). K/V tiles shared via
// double-buffered async LDS staging; K/V fragments pipelined behind WMMAs.
// ---------------------------------------------------------------------------
__global__ __launch_bounds__(256) void attn_kernel(
    const unsigned short* __restrict__ qbf, const unsigned short* __restrict__ kbf,
    const unsigned short* __restrict__ vt, const int* __restrict__ mask,
    float* __restrict__ out)
{
    __shared__ unsigned short Klds[2][32 * 64];    // 2 x 4 KB (row=key, 64 d)
    __shared__ unsigned short Vlds[2][64 * 32];    // 2 x 4 KB (row=d, 32 keys)
    __shared__ unsigned short Plds[8 * 512];       // per-wave 16x32 bf16 P tile
    const int tid  = threadIdx.x;
    const int lane = tid & 31;
    const int wave = tid >> 5;
    const int tile = blockIdx.x * 8 + wave;        // 0..1023
    const int b  = tile >> 7;
    const int qt = tile & 127;
    const int h  = lane >> 4;
    const int ln = lane & 15;
    unsigned short* pw = Plds + wave * 512;

    const unsigned short* kbase = kbf + (size_t)b * SEQ * DHEAD;
    const unsigned short* vbase = vt  + (size_t)b * DHEAD * SEQ;
    const int* mbase = mask + (size_t)b * SEQ * SEQ + (size_t)(qt * 16) * SEQ;

    // ---- Q A-fragments (16x64 = two 16x32 frags) ----
    ABf Qf[2];
    {
        const unsigned short* qp = qbf + (size_t)(b * SEQ + qt * 16 + ln) * DHEAD;
        #pragma unroll
        for (int j = 0; j < 2; ++j) {
            Qf[j].q[0] = *(const uint4*)(qp + j * 32 + h * 16);
            Qf[j].q[1] = *(const uint4*)(qp + j * 32 + h * 16 + 8);
        }
    }

    float m_[8], l_[8];
    v8f O[4];
    #pragma unroll
    for (int v = 0; v < 8; ++v) { m_[v] = -3.0e38f; l_[v] = 0.0f; }
    #pragma unroll
    for (int t = 0; t < 4; ++t)
        #pragma unroll
        for (int v = 0; v < 8; ++v) O[t][v] = 0.0f;

    // cooperative stage of one 32-key K tile + V tile (2 x 16B per thread)
    auto stageKV = [&](int kt, int buf) {
        int key0 = kt * 32;
        {   // K: 32 rows x 128B
            int row = tid >> 3, seg = tid & 7;
            cp16_g2l(kbase + (size_t)(key0 + row) * DHEAD + seg * 8,
                     &Klds[buf][row * 64 + seg * 8]);
        }
        {   // V: 64 rows x 64B
            int d = tid >> 2, seg = tid & 3;
            cp16_g2l(vbase + (size_t)d * SEQ + key0 + seg * 8,
                     &Vlds[buf][d * 32 + seg * 8]);
        }
    };

    stageKV(0, 0);
    int cur = 0;
    for (int kt = 0; kt < SEQ / 32; ++kt) {
        async_wait();
        __syncthreads();
        if (kt + 1 < SEQ / 32) stageKV(kt + 1, cur ^ 1);

        // prefetch next mask tile (global_prefetch_b8)
        if (kt + 1 < SEQ / 32)
            __builtin_prefetch(mbase + (size_t)(lane & 15) * SEQ
                               + (kt + 1) * 32 + (lane >> 4) * 16, 0, 0);

        // ---- S = (Q K^T)*scale + mask : pipelined K frags + 4 WMMA ----
        // fragment f = sub*2 + j : LDS row (sub*16+ln), d-offset j*32 + h*16
        v8f s[2];
        #pragma unroll
        for (int v = 0; v < 8; ++v) { s[0][v] = 0.0f; s[1][v] = 0.0f; }
        {
            ABf Kf[2];
            const unsigned short* k0 = &Klds[cur][ln * 64 + h * 16];
            Kf[0].q[0] = *(const uint4*)(k0);
            Kf[0].q[1] = *(const uint4*)(k0 + 8);
            #pragma unroll
            for (int f = 0; f < 4; ++f) {
                if (f + 1 < 4) {
                    int sub = (f + 1) >> 1, j = (f + 1) & 1;
                    const unsigned short* kp =
                        &Klds[cur][(sub * 16 + ln) * 64 + j * 32 + h * 16];
                    Kf[(f + 1) & 1].q[0] = *(const uint4*)(kp);
                    Kf[(f + 1) & 1].q[1] = *(const uint4*)(kp + 8);
                }
                s[f >> 1] = wmma_bf16(Qf[f & 1].v, Kf[f & 1].v, s[f >> 1]);
            }
        }

        #pragma unroll
        for (int sub = 0; sub < 2; ++sub) {
            const int* mp = mbase + kt * 32 + sub * 16 + ln;
            #pragma unroll
            for (int v = 0; v < 8; ++v) {
                float sv = s[sub][v] * 0.125f;               // 1/sqrt(64)
                int mk = mp[(size_t)(v + 8 * h) * SEQ];
                s[sub][v] = (mk == 0) ? -3.0e38f : sv;
            }
        }

        // ---- online softmax ----
        float mnew[8], alpha[8];
        #pragma unroll
        for (int v = 0; v < 8; ++v) {
            float r = fmaxf(s[0][v], s[1][v]);
            #pragma unroll
            for (int off = 1; off < 16; off <<= 1)
                r = fmaxf(r, __shfl_xor(r, off, 32));
            mnew[v]  = fmaxf(m_[v], r);
            alpha[v] = __expf(m_[v] - mnew[v]);
            m_[v]    = mnew[v];
        }
        #pragma unroll
        for (int v = 0; v < 8; ++v) {
            float p0 = __expf(s[0][v] - mnew[v]);
            float p1 = __expf(s[1][v] - mnew[v]);
            float r = p0 + p1;
            #pragma unroll
            for (int off = 1; off < 16; off <<= 1)
                r += __shfl_xor(r, off, 32);
            l_[v] = l_[v] * alpha[v] + r;
            pw[(v + 8 * h) * 32 + ln]      = f2bf(p0);   // C-layout -> LDS
            pw[(v + 8 * h) * 32 + 16 + ln] = f2bf(p1);
            #pragma unroll
            for (int t = 0; t < 4; ++t) O[t][v] *= alpha[v];
        }

        // ---- P A-fragment back from LDS (transposed layout) ----
        ABf Pf;
        Pf.q[0] = *(const uint4*)(pw + ln * 32 + h * 16);
        Pf.q[1] = *(const uint4*)(pw + ln * 32 + h * 16 + 8);

        // ---- O += P V : pipelined V frags + 4 WMMA ----
        {
            ABf Vf[2];
            const unsigned short* v0 = &Vlds[cur][ln * 32 + h * 16];
            Vf[0].q[0] = *(const uint4*)(v0);
            Vf[0].q[1] = *(const uint4*)(v0 + 8);
            #pragma unroll
            for (int t = 0; t < 4; ++t) {
                if (t + 1 < 4) {
                    const unsigned short* vp =
                        &Vlds[cur][((t + 1) * 16 + ln) * 32 + h * 16];
                    Vf[(t + 1) & 1].q[0] = *(const uint4*)(vp);
                    Vf[(t + 1) & 1].q[1] = *(const uint4*)(vp + 8);
                }
                O[t] = wmma_bf16(Pf.v, Vf[t & 1].v, O[t]);
            }
        }

        cur ^= 1;
    }

    // ---- epilogue ----
    float inv[8];
    #pragma unroll
    for (int v = 0; v < 8; ++v) inv[v] = 1.0f / (l_[v] + 1.0e-30f);
    float* op = out + (size_t)(b * SEQ + qt * 16) * DHEAD;
    #pragma unroll
    for (int t = 0; t < 4; ++t)
        #pragma unroll
        for (int v = 0; v < 8; ++v)
            op[(size_t)(v + 8 * h) * DHEAD + t * 16 + ln] = O[t][v] * inv[v];
}

// ---------------------------------------------------------------------------
extern "C" void kernel_launch(void* const* d_in, const int* in_sizes, int n_in,
                              void* d_out, int out_size, void* d_ws, size_t ws_size,
                              hipStream_t stream) {
    const float* x    = (const float*)d_in[0];
    const int*   mask = (const int*)  d_in[1];
    const float* Wq   = (const float*)d_in[2];
    const float* Wk   = (const float*)d_in[3];
    const float* Wv   = (const float*)d_in[4];
    float* out = (float*)d_out;

    char* ws = (char*)d_ws;
    unsigned short* wswz = (unsigned short*)(ws);                         // 384 KB
    unsigned short* qbf  = (unsigned short*)(ws + (512u << 10));          // 2 MB
    unsigned short* kbf  = (unsigned short*)(ws + (512u << 10) + (2u << 20));
    unsigned short* vt   = (unsigned short*)(ws + (512u << 10) + (4u << 20));

    wprep_kernel<<<768, 256, 0, stream>>>(Wq, Wk, Wv, wswz);
    qkv_kernel<<<128, 256, 0, stream>>>(x, wswz, qbf, kbf, vt);
    attn_kernel<<<128, 256, 0, stream>>>(qbf, kbf, vt, mask, out);
}
